// TwoLevelTransformerEncoder_73426760893119
// MI455X (gfx1250) — matrix operand order repaired and checked by reading, not compiled, not run on hardware
//
#include <hip/hip_runtime.h>
#include <hip/hip_bf16.h>

// ---------------------------------------------------------------------------
// Two-level transformer encoder for MI455X (gfx1250, wave32, WMMA).
// - All matmuls: tiled v_wmma_f32_16x16x32_bf16 GEMM, double-buffered LDS.
//   A tiles staged by the Tensor Data Mover (tensor_load_to_lds, TENSORcnt);
//   B tiles via registers (transposed store for single-load B fragments).
// - Attention: flash-attention on the same WMMA op; Q/K tiles staged with
//   global_load_async_to_lds_b128 (ASYNCcnt); softmax in C-fragment layout.
// ---------------------------------------------------------------------------

#define B_  2
#define T_  2048
#define D_  512
#define H_  8
#define DH_ 64
#define R_  8
#define DS_ 512
#define FF_ 2048
#define HP_ 2048
#define HR_ 2048
#define MF  (B_ * T_)        // 4096 fast tokens
#define KS_ (T_ / R_)        // 256 slow steps
#define MS  (B_ * KS_)       // 512 slow tokens

typedef __attribute__((ext_vector_type(8)))  __bf16 v8bf;
typedef __attribute__((ext_vector_type(16))) __bf16 v16bf;
typedef __attribute__((ext_vector_type(8)))  float  v8f;
typedef __attribute__((ext_vector_type(4)))  unsigned tdm_v4u;
typedef __attribute__((ext_vector_type(8)))  int      tdm_v8i;
typedef __attribute__((ext_vector_type(4)))  int      tdm_v4i;

#define WMMA_BF16(a, b, c) \
  __builtin_amdgcn_wmma_f32_16x16x32_bf16(false, (a), false, (b), (short)0, (c), false, false)

// Async DMA: global -> LDS, 16 bytes per active lane. DS address is the low
// 32 bits of the generic pointer.
__device__ __forceinline__ void async_copy_b128(void* lds_dst, const void* gsrc) {
  unsigned off = (unsigned)(unsigned long long)(uintptr_t)lds_dst;
  asm volatile("global_load_async_to_lds_b128 %0, %1, off"
               :: "v"(off), "v"(gsrc) : "memory");
}
__device__ __forceinline__ void wait_async0() {
  asm volatile("s_wait_asynccnt 0x0" ::: "memory");
}

// Tensor Data Mover: 2D bf16 tile (tile_k x tile_m) from row-major global
// [M_rows, K_elems] into LDS. Descriptor per CDNA5 ISA (08_async_tensor §8).
__device__ __forceinline__ void tdm_load_2d(const void* gsrc, void* lds_dst,
                                            unsigned tile_k, unsigned tile_m,
                                            unsigned K_elems, unsigned M_rows) {
  const unsigned long long ga = (unsigned long long)(uintptr_t)gsrc;
  const unsigned lds = (unsigned)(unsigned long long)(uintptr_t)lds_dst;
  tdm_v4u g0;
  g0[0] = 1u;                                                // count=1 (valid user D#)
  g0[1] = lds;                                               // lds_addr (bytes)
  g0[2] = (unsigned)(ga & 0xffffffffu);                      // global_addr[31:0]
  g0[3] = (unsigned)((ga >> 32) & 0x01ffffffu) | (2u << 30); // global_addr[56:32] | type=2
  tdm_v8i g1;
  g1[0] = (int)(1u << 16);                                   // data_size=2B
  g1[1] = (int)((K_elems & 0xffffu) << 16);                  // tensor_dim0[15:0]
  g1[2] = (int)((K_elems >> 16) | ((M_rows & 0xffffu) << 16)); // td0 hi | td1 lo
  g1[3] = (int)((M_rows >> 16) | (tile_k << 16));            // td1 hi | tile_dim0
  g1[4] = (int)tile_m;                                       // tile_dim1 (tile_dim2=0)
  g1[5] = (int)K_elems;                                      // tensor_dim0_stride[31:0]
  g1[6] = 0;                                                 // stride hi / dim1_stride lo
  g1[7] = 0;
  tdm_v4i z4 = (tdm_v4i){0, 0, 0, 0};
#if defined(__clang_major__) && (__clang_major__ >= 23)
  tdm_v8i z8 = (tdm_v8i){0, 0, 0, 0, 0, 0, 0, 0};
  __builtin_amdgcn_tensor_load_to_lds(g0, g1, z4, z4, z8, 0);
#else
  __builtin_amdgcn_tensor_load_to_lds(g0, g1, z4, z4, 0);
#endif
}

__device__ __forceinline__ float gelu_f(float x) {
  return 0.5f * x * (1.0f + tanhf(0.7978845608028654f * (x + 0.044715f * x * x * x)));
}

// 16x32 bf16 A-fragment from LDS (row-major, `pitch` elements/row).
// ISA: lanes 0-15 -> M=lane, K {0-7,16-23}; lanes 16-31 -> M=lane-16, K {8-15,24-31}.
__device__ __forceinline__ v16bf load_a_frag(const __bf16* base, int pitch) {
  const int lane = threadIdx.x & 31;
  const int m    = lane & 15;
  const int lo   = (lane < 16) ? 0 : 8;
  const __bf16* p = base + m * pitch + lo;
  v8bf x0 = *(const v8bf*)(p);
  v8bf x1 = *(const v8bf*)(p + 16);
  v16bf r;
#pragma unroll
  for (int i = 0; i < 8; ++i) { r[i] = x0[i]; r[i + 8] = x1[i]; }
  return r;
}

// 32x16 bf16 B-fragment; LDS holds K contiguous per column: bt[n*pitch + k].
__device__ __forceinline__ v16bf load_b_frag(const __bf16* bt, int pitch) {
  const int lane = threadIdx.x & 31;
  const int n    = lane & 15;
  const int kb   = (lane < 16) ? 0 : 8;
  const __bf16* p = bt + n * pitch + kb;
  v8bf x0 = *(const v8bf*)(p);
  v8bf x1 = *(const v8bf*)(p + 16);
  v16bf r;
#pragma unroll
  for (int i = 0; i < 8; ++i) { r[i] = x0[i]; r[i + 8] = x1[i]; }
  return r;
}

// ---------------------------------------------------------------------------
// GEMM: C[M,N] = epi(A[M,K]_bf16 @ B[K,N]_bf16 + bias) (+res), f32/bf16 out.
// 256 threads = 8 waves; tile 128x128, K step 32; wave tile 32x64.
// Double-buffered LDS; A staged via TDM (wave 0), B via regs.
// ---------------------------------------------------------------------------
template <bool HAS_BIAS, bool ACT_GELU, bool HAS_RES, bool WF32, bool WB16>
__global__ __launch_bounds__(256)
void gemm_wmma_k(const __bf16* __restrict__ A, const __bf16* __restrict__ Bm,
                 const float* __restrict__ bias, const float* __restrict__ res,
                 float* __restrict__ Cf, __bf16* __restrict__ Cb,
                 int M, int N, int K) {
  __shared__ __align__(32) __bf16 As[2][128 * 32];   // [m][k]
  __shared__ __align__(32) __bf16 Bt[2][128 * 32];   // [n][k]

  const int tid   = threadIdx.x;
  const int lane  = tid & 31;
  const int wave  = tid >> 5;
  const int wm    = wave & 3;   // 4 waves along M, 32 rows each
  const int wn    = wave >> 2;  // 2 waves along N, 64 cols each
  const int mBase = blockIdx.y * 128;
  const int nBase = blockIdx.x * 128;

  // per-thread chunk coordinates for the B tile (two 8-element chunks)
  const int c1  = tid + 256;
  const int bk0 = tid >> 4, bn0 = (tid & 15) << 3;
  const int bk1 = c1 >> 4,  bn1 = (c1 & 15) << 3;

  v8bf bv0, bv1;
  auto stage_A = [&](int bufi, int k0) {    // one TDM descriptor per slab
    if (wave == 0)
      tdm_load_2d(A + (size_t)mBase * K + k0, &As[bufi][0], 32u, 128u,
                  (unsigned)K, (unsigned)M);
  };
  auto load_B = [&](int k0) {
    bv0 = *(const v8bf*)(Bm + (size_t)(k0 + bk0) * N + nBase + bn0);
    bv1 = *(const v8bf*)(Bm + (size_t)(k0 + bk1) * N + nBase + bn1);
  };
  auto store_B = [&](int bufi) {
#pragma unroll
    for (int j = 0; j < 8; ++j) Bt[bufi][(bn0 + j) * 32 + bk0] = bv0[j];
#pragma unroll
    for (int j = 0; j < 8; ++j) Bt[bufi][(bn1 + j) * 32 + bk1] = bv1[j];
  };

  v8f acc[2][4];
#pragma unroll
  for (int mi = 0; mi < 2; ++mi)
#pragma unroll
    for (int ni = 0; ni < 4; ++ni) acc[mi][ni] = (v8f){0.f,0.f,0.f,0.f,0.f,0.f,0.f,0.f};

  // prologue: stage slab 0
  stage_A(0, 0);
  load_B(0);
  store_B(0);
  if (wave == 0) __builtin_amdgcn_s_wait_tensorcnt(0);
  __syncthreads();

  int buf = 0;
  for (int k0 = 0; k0 < K; k0 += 32) {
    const int  nxt      = buf ^ 1;
    const bool has_next = (k0 + 32 < K);
    if (has_next) {                 // kick next slab while this one computes
      stage_A(nxt, k0 + 32);
      load_B(k0 + 32);
      if (k0 + 64 < K)              // prefetch B slab after next
        __builtin_prefetch(Bm + (size_t)(k0 + 64 + bk0) * N + nBase + bn0, 0, 3);
    }

    v16bf af[2];
    af[0] = load_a_frag(&As[buf][(wm * 32 + 0)  * 32], 32);
    af[1] = load_a_frag(&As[buf][(wm * 32 + 16) * 32], 32);
    v16bf bfr[4];
#pragma unroll
    for (int ni = 0; ni < 4; ++ni)
      bfr[ni] = load_b_frag(&Bt[buf][(wn * 64 + ni * 16) * 32], 32);
#pragma unroll
    for (int mi = 0; mi < 2; ++mi)
#pragma unroll
      for (int ni = 0; ni < 4; ++ni)
        acc[mi][ni] = WMMA_BF16(af[mi], bfr[ni], acc[mi][ni]);

    if (has_next) store_B(nxt);
    if (wave == 0) __builtin_amdgcn_s_wait_tensorcnt(0);  // own DMA done
    __syncthreads();
    buf = nxt;
  }

  // epilogue (C layout: VGPR i -> row i (lanes 0-15) / i+8 (lanes 16-31), col = lane&15)
  const int hi8 = (lane >> 4) << 3;
  const int cn  = lane & 15;
#pragma unroll
  for (int mi = 0; mi < 2; ++mi) {
#pragma unroll
    for (int ni = 0; ni < 4; ++ni) {
      const int gcol = nBase + wn * 64 + ni * 16 + cn;
      const float bia = HAS_BIAS ? bias[gcol] : 0.0f;
#pragma unroll
      for (int i = 0; i < 8; ++i) {
        const int grow = mBase + wm * 32 + mi * 16 + i + hi8;
        float v = acc[mi][ni][i] + bia;
        if (ACT_GELU) v = gelu_f(v);
        if (HAS_RES)  v += res[(size_t)grow * N + gcol];
        if (WF32) Cf[(size_t)grow * N + gcol] = v;
        if (WB16) Cb[(size_t)grow * N + gcol] = (__bf16)v;
      }
    }
  }
}

// ---------------------------------------------------------------------------
// LayerNorm over last dim (512), f32 in -> bf16 out (gamma/beta applied).
// ---------------------------------------------------------------------------
__global__ __launch_bounds__(256)
void layernorm_k(const float* __restrict__ x, const float* __restrict__ g,
                 const float* __restrict__ bta, __bf16* __restrict__ out, int Dd) {
  const int row = blockIdx.x;
  const float* xr = x + (size_t)row * Dd;
  __shared__ float sred[8], ssred[8], stats[2];
  float s = 0.f, ss = 0.f;
  for (int i = threadIdx.x; i < Dd; i += 256) { float v = xr[i]; s += v; ss += v * v; }
#pragma unroll
  for (int o = 16; o > 0; o >>= 1) { s += __shfl_down(s, o, 32); ss += __shfl_down(ss, o, 32); }
  if ((threadIdx.x & 31) == 0) { sred[threadIdx.x >> 5] = s; ssred[threadIdx.x >> 5] = ss; }
  __syncthreads();
  if (threadIdx.x == 0) {
    float ts = 0.f, tss = 0.f;
    for (int w = 0; w < 8; ++w) { ts += sred[w]; tss += ssred[w]; }
    float mean = ts / Dd;
    stats[0] = mean;
    stats[1] = rsqrtf(tss / Dd - mean * mean + 1e-5f);
  }
  __syncthreads();
  const float mean = stats[0], rstd = stats[1];
  for (int i = threadIdx.x; i < Dd; i += 256)
    out[(size_t)row * Dd + i] = (__bf16)((xr[i] - mean) * rstd * g[i] + bta[i]);
}

// ---------------------------------------------------------------------------
// Flash attention, causal, dh=64. One wave per 16-query tile per (b,h).
// qkv bf16 [B*T, 3*Dm]: q @ h*64, k @ Dm + h*64, v @ 2*Dm + h*64.
// ---------------------------------------------------------------------------
__global__ __launch_bounds__(32)
void attn_k(const __bf16* __restrict__ qkv, __bf16* __restrict__ out, int T, int Dm) {
  const int qbase = blockIdx.x * 16;
  const int h     = blockIdx.y;
  const int b     = blockIdx.z;
  const int lane  = threadIdx.x;
  const int hi    = lane >> 4;     // half-wave id
  const int ln16  = lane & 15;
  const int threeD = 3 * Dm;

  __shared__ __align__(32) __bf16 Qs[16 * DH_];
  __shared__ __align__(32) __bf16 Ks[32 * DH_];     // [key][d]
  __shared__ __align__(32) __bf16 Vt[DH_ * 32];     // [d][key] (transposed)
  __shared__ __align__(32) __bf16 Ps[16 * 32];      // P tile for PV wmma

  { // load Q tile 16x64 via async DMA
    const int r  = lane >> 1;
    const int c0 = (lane & 1) * 32;
    const __bf16* src = qkv + (size_t)(b * T + qbase + r) * threeD + h * DH_ + c0;
#pragma unroll
    for (int j = 0; j < 32; j += 8)
      async_copy_b128(Qs + r * DH_ + c0 + j, src + j);
  }
  wait_async0();
  __syncthreads();
  const v16bf qa0 = load_a_frag(Qs + 0,  DH_);   // d 0..31
  const v16bf qa1 = load_a_frag(Qs + 32, DH_);   // d 32..63

  v8f ofr[4];
#pragma unroll
  for (int t = 0; t < 4; ++t) ofr[t] = (v8f){0.f,0.f,0.f,0.f,0.f,0.f,0.f,0.f};
  float mrow[8], lrow[8];
#pragma unroll
  for (int i = 0; i < 8; ++i) { mrow[i] = -1e30f; lrow[i] = 0.f; }

  for (int kb = 0; kb < qbase + 16; kb += 32) {
    { // load K (async DMA) and V (transposed, via regs); one key row per lane
      const int key = kb + lane;
      if (key < T) {
        const __bf16* kp = qkv + (size_t)(b * T + key) * threeD + Dm + h * DH_;
        const __bf16* vp = kp + Dm;
#pragma unroll
        for (int j = 0; j < DH_; j += 8)
          async_copy_b128(Ks + lane * DH_ + j, kp + j);
#pragma unroll
        for (int j = 0; j < DH_; ++j) Vt[j * 32 + lane] = vp[j];
      } else {
#pragma unroll
        for (int j = 0; j < DH_; ++j) { Ks[lane * DH_ + j] = (__bf16)0.f; Vt[j * 32 + lane] = (__bf16)0.f; }
      }
    }
    wait_async0();
    __syncthreads();

    // S = Q K^T for two 16-key halves (B = K^T: column n = key row of Ks)
    v8f sfr[2];
#pragma unroll
    for (int nt = 0; nt < 2; ++nt) {
      v8f s = (v8f){0.f,0.f,0.f,0.f,0.f,0.f,0.f,0.f};
      s = WMMA_BF16(qa0, load_b_frag(Ks + (nt * 16) * DH_ + 0,  DH_), s);
      s = WMMA_BF16(qa1, load_b_frag(Ks + (nt * 16) * DH_ + 32, DH_), s);
      sfr[nt] = s;
    }

    // scale + causal mask + online softmax (row = i + hi*8, col = lane&15)
    float tmax[8];
#pragma unroll
    for (int i = 0; i < 8; ++i) {
      const int row = qbase + i + hi * 8;
      float a0 = sfr[0][i] * 0.125f; const int c0 = kb + ln16;
      float a1 = sfr[1][i] * 0.125f; const int c1 = kb + 16 + ln16;
      if (c0 > row) a0 = -1e30f;
      if (c1 > row) a1 = -1e30f;
      sfr[0][i] = a0; sfr[1][i] = a1;
      float mx = fmaxf(a0, a1);
#pragma unroll
      for (int o = 1; o < 16; o <<= 1) mx = fmaxf(mx, __shfl_xor(mx, o, 32));
      tmax[i] = mx;
    }
#pragma unroll
    for (int i = 0; i < 8; ++i) {
      const float mnew = fmaxf(mrow[i], tmax[i]);
      const float corr = __expf(mrow[i] - mnew);
      const float p0 = __expf(sfr[0][i] - mnew);
      const float p1 = __expf(sfr[1][i] - mnew);
      float ssum = p0 + p1;
#pragma unroll
      for (int o = 1; o < 16; o <<= 1) ssum += __shfl_xor(ssum, o, 32);
      lrow[i] = lrow[i] * corr + ssum;
      mrow[i] = mnew;
#pragma unroll
      for (int t = 0; t < 4; ++t) ofr[t][i] *= corr;
      Ps[(i + hi * 8) * 32 + ln16]      = (__bf16)p0;
      Ps[(i + hi * 8) * 32 + 16 + ln16] = (__bf16)p1;
    }
    __syncthreads();

    // O += P V  (P: 16x32 A frag; V^T gives B frags per 16-wide d tile)
    const v16bf pa = load_a_frag(Ps, 32);
#pragma unroll
    for (int nt = 0; nt < 4; ++nt)
      ofr[nt] = WMMA_BF16(pa, load_b_frag(Vt + (nt * 16) * 32, 32), ofr[nt]);
    __syncthreads();
  }

  // normalize + store bf16
#pragma unroll
  for (int i = 0; i < 8; ++i) {
    const size_t grow = (size_t)(b * T + qbase + i + hi * 8);
    const float inv = 1.0f / lrow[i];
#pragma unroll
    for (int nt = 0; nt < 4; ++nt)
      out[grow * Dm + h * DH_ + nt * 16 + ln16] = (__bf16)(ofr[nt][i] * inv);
  }
}

// ---------------------------------------------------------------------------
// Glue kernels
// ---------------------------------------------------------------------------
__global__ void cvt_k(const float* __restrict__ in, __bf16* __restrict__ out, int n) {
  const int i = blockIdx.x * 256 + threadIdx.x;
  if (i < n) out[i] = (__bf16)in[i];
}

// causal pool shift: z[b,0,:]=0, z[b,k,:]=s[b,k-1,:]
__global__ void shift_k(const float* __restrict__ s, float* __restrict__ z,
                        int Kn, int Dn) {
  const int i = blockIdx.x * 256 + threadIdx.x;
  const int total = B_ * Kn * Dn;
  if (i >= total) return;
  const int d = i % Dn;
  const int k = (i / Dn) % Kn;
  const int bb = i / (Dn * Kn);
  z[i] = (k == 0) ? 0.f : s[((size_t)bb * Kn + (k - 1)) * Dn + d];
}

// concat [zf | repeat_interleave(zs, R)] -> bf16 [B*T, D+DS]
__global__ void concat_k(const __bf16* __restrict__ zf, const __bf16* __restrict__ zs,
                         __bf16* __restrict__ cat, int Tt, int Dd, int Dss, int Rr) {
  const int i = blockIdx.x * 256 + threadIdx.x;
  const int W = Dd + Dss;
  const int total = B_ * Tt * W;
  if (i >= total) return;
  const int c = i % W;
  const int row = i / W;
  const int bb = row / Tt;
  const int t = row - bb * Tt;
  cat[i] = (c < Dd) ? zf[(size_t)row * Dd + c]
                    : zs[((size_t)(bb * (Tt / Rr) + t / Rr)) * Dss + (c - Dd)];
}

// ---------------------------------------------------------------------------
// Host orchestration
// ---------------------------------------------------------------------------
static void encoder_layer(hipStream_t st, const float* xin, int Mrows, int Tt,
                          const float* ln1g, const float* ln1b,
                          const __bf16* wqkv, const float* bqkv,
                          const __bf16* wo, const float* bo,
                          const float* ln2g, const float* ln2b,
                          const __bf16* w1, const float* b1,
                          const __bf16* w2, const float* b2,
                          __bf16* h_buf, __bf16* qkv_buf, __bf16* attn_buf,
                          float* x1_buf, __bf16* t1_buf, __bf16* out_b16) {
  layernorm_k<<<Mrows, 256, 0, st>>>(xin, ln1g, ln1b, h_buf, D_);
  gemm_wmma_k<true, false, false, false, true>
      <<<dim3((3 * D_) / 128, Mrows / 128), 256, 0, st>>>(
          h_buf, wqkv, bqkv, nullptr, nullptr, qkv_buf, Mrows, 3 * D_, D_);
  attn_k<<<dim3(Tt / 16, H_, B_), 32, 0, st>>>(qkv_buf, attn_buf, Tt, D_);
  gemm_wmma_k<true, false, true, true, false>
      <<<dim3(D_ / 128, Mrows / 128), 256, 0, st>>>(
          attn_buf, wo, bo, xin, x1_buf, nullptr, Mrows, D_, D_);
  layernorm_k<<<Mrows, 256, 0, st>>>(x1_buf, ln2g, ln2b, h_buf, D_);
  gemm_wmma_k<true, true, false, false, true>
      <<<dim3(FF_ / 128, Mrows / 128), 256, 0, st>>>(
          h_buf, w1, b1, nullptr, nullptr, t1_buf, Mrows, FF_, D_);
  gemm_wmma_k<true, false, true, false, true>
      <<<dim3(D_ / 128, Mrows / 128), 256, 0, st>>>(
          t1_buf, w2, b2, x1_buf, nullptr, out_b16, Mrows, D_, FF_);
}

extern "C" void kernel_launch(void* const* d_in, const int* in_sizes, int n_in,
                              void* d_out, int out_size, void* d_ws, size_t ws_size,
                              hipStream_t stream) {
  const float* x      = (const float*)d_in[0];
  const float* ln1f_g = (const float*)d_in[1];
  const float* ln1f_b = (const float*)d_in[2];
  const float* wqkv_f = (const float*)d_in[3];
  const float* bqkv_f = (const float*)d_in[4];
  const float* wo_f   = (const float*)d_in[5];
  const float* bo_f   = (const float*)d_in[6];
  const float* ln2f_g = (const float*)d_in[7];
  const float* ln2f_b = (const float*)d_in[8];
  const float* w1_f   = (const float*)d_in[9];
  const float* b1_f   = (const float*)d_in[10];
  const float* w2_f   = (const float*)d_in[11];
  const float* b2_f   = (const float*)d_in[12];
  const float* ln1s_g = (const float*)d_in[13];
  const float* ln1s_b = (const float*)d_in[14];
  const float* wqkv_s = (const float*)d_in[15];
  const float* bqkv_s = (const float*)d_in[16];
  const float* wo_s   = (const float*)d_in[17];
  const float* bo_s   = (const float*)d_in[18];
  const float* ln2s_g = (const float*)d_in[19];
  const float* ln2s_b = (const float*)d_in[20];
  const float* w1_s   = (const float*)d_in[21];
  const float* b1_s   = (const float*)d_in[22];
  const float* w2_s   = (const float*)d_in[23];
  const float* b2_s   = (const float*)d_in[24];
  const float* wp1    = (const float*)d_in[25];
  const float* bp1    = (const float*)d_in[26];
  const float* wp2    = (const float*)d_in[27];
  const float* bp2    = (const float*)d_in[28];
  const float* wr1    = (const float*)d_in[29];
  const float* br1    = (const float*)d_in[30];
  const float* wr2    = (const float*)d_in[31];
  const float* br2    = (const float*)d_in[32];

  char* ws = (char*)d_ws;
  auto alloc = [&](size_t bytes) -> char* {
    char* r = ws; ws += (bytes + 255) & ~(size_t)255; return r;
  };
  auto A2 = [&](size_t n) -> __bf16* { return (__bf16*)alloc(n * sizeof(__bf16)); };
  auto A4 = [&](size_t n) -> float*  { return (float*)alloc(n * sizeof(float)); };

  // bf16 weight copies
  __bf16* wqkvf_b = A2((size_t)D_ * 3 * D_);
  __bf16* wof_b   = A2((size_t)D_ * D_);
  __bf16* w1f_b   = A2((size_t)D_ * FF_);
  __bf16* w2f_b   = A2((size_t)FF_ * D_);
  __bf16* wqkvs_b = A2((size_t)DS_ * 3 * DS_);
  __bf16* wos_b   = A2((size_t)DS_ * DS_);
  __bf16* w1s_b   = A2((size_t)DS_ * FF_);
  __bf16* w2s_b   = A2((size_t)FF_ * DS_);
  __bf16* wp1_b   = A2((size_t)R_ * D_ * HP_);
  __bf16* wp2_b   = A2((size_t)HP_ * DS_);
  __bf16* wr1_b   = A2((size_t)(D_ + DS_) * HR_);
  __bf16* wr2_b   = A2((size_t)HR_ * D_);
  // activations (slow encoder reuses the fast buffers)
  __bf16* h_buf    = A2((size_t)MF * D_);
  __bf16* qkv_buf  = A2((size_t)MF * 3 * D_);
  __bf16* attn_buf = A2((size_t)MF * D_);
  float*  x1_buf   = A4((size_t)MF * D_);
  __bf16* t1_buf   = A2((size_t)MF * FF_);     // reused for pool hidden and r1
  __bf16* zf_b16   = A2((size_t)MF * D_);
  float*  sfull    = A4((size_t)MS * DS_);
  float*  zsin     = A4((size_t)MS * DS_);
  __bf16* zs_b16   = A2((size_t)MS * DS_);
  __bf16* cat_b16  = A2((size_t)MF * (D_ + DS_));

  auto cvt = [&](const float* src, __bf16* dst, int n) {
    cvt_k<<<(n + 255) / 256, 256, 0, stream>>>(src, dst, n);
  };
  cvt(wqkv_f, wqkvf_b, D_ * 3 * D_);
  cvt(wo_f,   wof_b,   D_ * D_);
  cvt(w1_f,   w1f_b,   D_ * FF_);
  cvt(w2_f,   w2f_b,   FF_ * D_);
  cvt(wqkv_s, wqkvs_b, DS_ * 3 * DS_);
  cvt(wo_s,   wos_b,   DS_ * DS_);
  cvt(w1_s,   w1s_b,   DS_ * FF_);
  cvt(w2_s,   w2s_b,   FF_ * DS_);
  cvt(wp1,    wp1_b,   R_ * D_ * HP_);
  cvt(wp2,    wp2_b,   HP_ * DS_);
  cvt(wr1,    wr1_b,   (D_ + DS_) * HR_);
  cvt(wr2,    wr2_b,   HR_ * D_);

  // ---- fast encoder layer
  encoder_layer(stream, x, MF, T_, ln1f_g, ln1f_b, wqkvf_b, bqkv_f, wof_b, bo_f,
                ln2f_g, ln2f_b, w1f_b, b1_f, w2f_b, b2_f,
                h_buf, qkv_buf, attn_buf, x1_buf, t1_buf, zf_b16);

  // ---- causal pool: zf viewed as [MS, R*D]
  gemm_wmma_k<true, true, false, false, true>
      <<<dim3(HP_ / 128, MS / 128), 256, 0, stream>>>(
          zf_b16, wp1_b, bp1, nullptr, nullptr, t1_buf, MS, HP_, R_ * D_);
  gemm_wmma_k<true, false, false, true, false>
      <<<dim3(DS_ / 128, MS / 128), 256, 0, stream>>>(
          t1_buf, wp2_b, bp2, nullptr, sfull, nullptr, MS, DS_, HP_);
  shift_k<<<(MS * DS_ + 255) / 256, 256, 0, stream>>>(sfull, zsin, KS_, DS_);

  // ---- slow encoder layer
  encoder_layer(stream, zsin, MS, KS_, ln1s_g, ln1s_b, wqkvs_b, bqkv_s, wos_b, bo_s,
                ln2s_g, ln2s_b, w1s_b, b1_s, w2s_b, b2_s,
                h_buf, qkv_buf, attn_buf, x1_buf, t1_buf, zs_b16);

  // ---- recombine
  concat_k<<<(MF * (D_ + DS_) + 255) / 256, 256, 0, stream>>>(
      zf_b16, zs_b16, cat_b16, T_, D_, DS_, R_);
  gemm_wmma_k<true, true, false, false, true>
      <<<dim3(HR_ / 128, MF / 128), 256, 0, stream>>>(
          cat_b16, wr1_b, br1, nullptr, nullptr, t1_buf, MF, HR_, D_ + DS_);
  gemm_wmma_k<true, false, false, true, false>
      <<<dim3(D_ / 128, MF / 128), 256, 0, stream>>>(
          t1_buf, wr2_b, br2, nullptr, (float*)d_out, nullptr, MF, D_, HR_);
}
